// AreaAttentionWrapper_90280212562346
// MI455X (gfx1250) — matrix operand order repaired and checked by reading, not compiled, not run on hardware
//
#include <hip/hip_runtime.h>
#include <hip/hip_bf16.h>

// ---------------------------------------------------------------------------
// Area attention for MI455X (gfx1250, wave32, WMMA f32_16x16x32_f16,
// async global->LDS copies with ASYNCcnt double buffering when available).
//
// Pipeline:
//   0. convert x -> f16, w_o -> f16 transposed
//   1. combine_kernel:  WcT_p = (w_qkv[:,p] @ w_p)^T  stored f16 [n][k]
//   2. gemm_wmma<0> x3: qh/kh/vh = x_h @ WcT_p^T + b_p -> [BH, N, 64] f32
//   3. integral_kernel: 2D summed-area tables I_k, I_v [BH, 17, 17, 64] f32
//   4. area_attn_kernel: fused area pooling + flash softmax + WMMA, out f16
//   5. gemm_wmma<1>: out = attn_h @ WoT^T + b_o  (f32)
// ---------------------------------------------------------------------------

typedef __attribute__((ext_vector_type(16))) _Float16 v16h;
typedef __attribute__((ext_vector_type(8)))  float    v8f;
typedef __attribute__((ext_vector_type(4)))  int      v4i;

#define DIMX   768
#define NTOK   256
#define BATCH  16
#define HEADS  12
#define DHEAD  64
#define BHX    (BATCH*HEADS)     // 192
#define MROWS  (BATCH*NTOK)      // 4096
#define NAREA  2025
#define ISZ    (17*17*64)        // 18496 floats per (b,h) integral image

#ifndef HAVE_ASYNC_LDS
#if defined(__has_builtin)
#if __has_builtin(__builtin_amdgcn_global_load_async_to_lds_b128) && \
    __has_builtin(__builtin_amdgcn_s_wait_asynccnt)
#define HAVE_ASYNC_LDS 1
#else
#define HAVE_ASYNC_LDS 0
#endif
#else
#define HAVE_ASYNC_LDS 0
#endif
#endif

#if HAVE_ASYNC_LDS
typedef v4i __attribute__((address_space(1)))* gv4i_p;   // global int4*
typedef v4i __attribute__((address_space(3)))* lv4i_p;   // LDS int4*
__device__ __forceinline__ void async_b128(const void* g, void* l) {
    __builtin_amdgcn_global_load_async_to_lds_b128(
        (gv4i_p)const_cast<void*>(g), (lv4i_p)l, /*offset=*/0, /*cpol=*/0);
}
#define ASYNC_WAIT(n) __builtin_amdgcn_s_wait_asynccnt(n)
#endif

union Frag16 { v16h v; unsigned int u[8]; _Float16 h[16]; };

// CDNA5 16-bit A-matrix 16x32 layout: VGPR i (i<4) covers K=2i,2i+1 ;
// i>=4 covers K=16+2(i-4),.. ; lanes 16-31 add K offset +8.
__device__ __forceinline__ int kbaseA(int i) {
    return (i < 4) ? (2 * i) : (16 + 2 * (i - 4));
}

__device__ __forceinline__ v8f wmma_f16(v16h a, v16h b, v8f c) {
    return __builtin_amdgcn_wmma_f32_16x16x32_f16(
        false, a, false, b, (short)0, c, false, false);
}

__device__ __forceinline__ float rowmax16(float x) {
    x = fmaxf(x, __shfl_xor(x, 1, 16));
    x = fmaxf(x, __shfl_xor(x, 2, 16));
    x = fmaxf(x, __shfl_xor(x, 4, 16));
    x = fmaxf(x, __shfl_xor(x, 8, 16));
    return x;
}
__device__ __forceinline__ float rowsum16(float x) {
    x += __shfl_xor(x, 1, 16);
    x += __shfl_xor(x, 2, 16);
    x += __shfl_xor(x, 4, 16);
    x += __shfl_xor(x, 8, 16);
    return x;
}

// ---------------------------------------------------------------------------
// 0a. f32 -> f16 vectorized convert (n4 = quads)
// ---------------------------------------------------------------------------
__global__ __launch_bounds__(256)
void convert_f16_kernel(const float* __restrict__ in, _Float16* __restrict__ out,
                        int n4) {
    int t = blockIdx.x * 256 + threadIdx.x;
    if (t < n4) {
        float4 v = ((const float4*)in)[t];
        union { _Float16 h[4]; uint2 u; } pk;
        pk.h[0] = (_Float16)v.x; pk.h[1] = (_Float16)v.y;
        pk.h[2] = (_Float16)v.z; pk.h[3] = (_Float16)v.w;
        ((uint2*)out)[t] = pk.u;
    }
}

// ---------------------------------------------------------------------------
// 0b. w_o [k][n] f32 -> WoT [n][k] f16
// ---------------------------------------------------------------------------
__global__ __launch_bounds__(256)
void transpose_f16_kernel(const float* __restrict__ in, _Float16* __restrict__ out) {
    int t = blockIdx.x * 256 + threadIdx.x;      // < 768*768
    int k = t / DIMX, n = t - k * DIMX;
    out[(long)n * DIMX + k] = (_Float16)in[t];
}

// ---------------------------------------------------------------------------
// 1. WcT[p][n][i] = sum_j w_qkv[i][p*768+j] * w_p[j][n]   (f16, transposed)
// ---------------------------------------------------------------------------
__global__ __launch_bounds__(256)
void combine_kernel(const float* __restrict__ w_qkv,
                    const float* __restrict__ w_q,
                    const float* __restrict__ w_k,
                    const float* __restrict__ w_v,
                    _Float16* __restrict__ WcT) {
    int t = blockIdx.x * 256 + threadIdx.x;      // < 3*768*768
    int p   = t / (DIMX * DIMX);
    int rem = t - p * (DIMX * DIMX);
    int i   = rem / DIMX;
    int n   = rem - i * DIMX;
    const float* wp = (p == 0) ? w_q : (p == 1) ? w_k : w_v;
    const float* a  = w_qkv + (long)i * (3 * DIMX) + p * DIMX;
    float acc = 0.0f;
    for (int j = 0; j < DIMX; ++j)
        acc += a[j] * wp[(long)j * DIMX + n];
    WcT[(long)p * DIMX * DIMX + (long)n * DIMX + i] = (_Float16)acc;
}

// ---------------------------------------------------------------------------
// 2./5. WMMA GEMM: out = A[4096x768]f16 @ BT[768x768]^T f16 + bias
//   MODE 0: scatter to per-head layout [(b*12+h)*256+nt, d]  (f32)
//   MODE 1: plain row-major [row, col]                       (f32)
// Block: 128 threads (4 waves); tile 64(M) x 64(N); K stepped by 32;
// async global->LDS with 2-deep double buffering when available.
// ---------------------------------------------------------------------------
template <int MODE>
__global__ __launch_bounds__(128)
void gemm_wmma_kernel(const _Float16* __restrict__ A,
                      const _Float16* __restrict__ BT,
                      const float* __restrict__ bias,
                      float* __restrict__ out) {
    __shared__ __align__(16) _Float16 alds[2][64 * 32];   // [buf][m][k]
    __shared__ __align__(16) _Float16 blds[2][64 * 32];   // [buf][n][k]

    const int tid  = threadIdx.x;
    const int lane = tid & 31;
    const int wave = tid >> 5;
    const int hf   = lane >> 4;
    const int ln   = lane & 15;
    const int col0 = blockIdx.x * 64;
    const int rm0  = blockIdx.y * 64;

    v8f acc[4];
#pragma unroll
    for (int c = 0; c < 4; ++c)
#pragma unroll
        for (int r = 0; r < 8; ++r) acc[c][r] = 0.0f;

    // 256 16-byte chunks per matrix per tile; this thread moves chunks tid,
    // tid+128. Chunk c -> row m = c>>2, k-offset = (c&3)*8 halves.
    auto issue_tile = [&](int k0, int buf) {
#pragma unroll
        for (int cc = 0; cc < 2; ++cc) {
            const int ch = tid + cc * 128;
            const int m  = ch >> 2;
            const int ko = (ch & 3) * 8;
            const _Float16* ga = A  + (long)(rm0  + m) * DIMX + k0 + ko;
            const _Float16* gb = BT + (long)(col0 + m) * DIMX + k0 + ko;
            _Float16* la = &alds[buf][m * 32 + ko];
            _Float16* lb = &blds[buf][m * 32 + ko];
#if HAVE_ASYNC_LDS
            async_b128(ga, la);
            async_b128(gb, lb);
#else
            *(uint4*)la = *(const uint4*)ga;
            *(uint4*)lb = *(const uint4*)gb;
#endif
        }
    };

    auto compute_tile = [&](int buf) {
        Frag16 af;
        const int m = wave * 16 + ln;
#pragma unroll
        for (int i = 0; i < 8; ++i) {
            int kk = kbaseA(i) + 8 * hf;
            af.u[i] = *(const unsigned int*)&alds[buf][m * 32 + kk];
        }
#pragma unroll
        for (int c = 0; c < 4; ++c) {
            Frag16 bf;
            const int n = c * 16 + ln;
#pragma unroll
            for (int i = 0; i < 8; ++i) {
                int kk = kbaseA(i) + 8 * hf;
                bf.u[i] = *(const unsigned int*)&blds[buf][n * 32 + kk];
            }
            acc[c] = wmma_f16(af.v, bf.v, acc[c]);
        }
    };

    const int KITER = DIMX / 32;   // 24
#if HAVE_ASYNC_LDS
    issue_tile(0, 0);
    for (int it = 0; it < KITER; ++it) {
        if (it + 1 < KITER) {
            issue_tile((it + 1) * 32, (it + 1) & 1);
            ASYNC_WAIT(4);     // oldest 4 (tile `it`) complete, in-order
        } else {
            ASYNC_WAIT(0);
        }
        __syncthreads();
        compute_tile(it & 1);
        __syncthreads();
    }
#else
    for (int it = 0; it < KITER; ++it) {
        issue_tile(it * 32, 0);
        __syncthreads();
        compute_tile(0);
        __syncthreads();
    }
#endif

    // --- epilogue: bias + scatter ---
#pragma unroll
    for (int c = 0; c < 4; ++c)
#pragma unroll
        for (int r = 0; r < 8; ++r) {
            int row = rm0 + wave * 16 + r + 8 * hf;
            int col = col0 + c * 16 + ln;
            float val = acc[c][r] + bias[col];
            if constexpr (MODE == 0) {
                int b = row >> 8, nt = row & 255;
                int h = col >> 6, d = col & 63;
                out[(((long)(b * HEADS + h) * NTOK) + nt) * DHEAD + d] = val;
            } else {
                out[(long)row * DIMX + col] = val;
            }
        }
}

// ---------------------------------------------------------------------------
// 3. Integral images with zero padding row/col: I[bh][17][17][64]
// ---------------------------------------------------------------------------
__global__ __launch_bounds__(256)
void integral_kernel(const float* __restrict__ kh, const float* __restrict__ vh,
                     float* __restrict__ Ik, float* __restrict__ Iv) {
    const int bh = blockIdx.x;
    const float* src = (blockIdx.y ? vh : kh) + (long)bh * (NTOK * DHEAD);
    float* dst = (blockIdx.y ? Iv : Ik) + (long)bh * ISZ;
    const int tid = threadIdx.x;

    for (int i = tid; i < 2112; i += 256) {
        int y, x, d;
        if (i < 1088) { y = 0; x = i >> 6; d = i & 63; }
        else { int j = i - 1088; y = 1 + (j >> 6); x = 0; d = j & 63; }
        dst[(y * 17 + x) * 64 + d] = 0.0f;
    }
    for (int t = tid; t < 1024; t += 256) {
        int h = t >> 6, d = t & 63;
        float run = 0.0f;
        for (int w = 0; w < 16; ++w) {
            run += src[(h * 16 + w) * 64 + d];
            dst[((h + 1) * 17 + (w + 1)) * 64 + d] = run;
        }
    }
    __threadfence_block();
    __syncthreads();
    for (int t = tid; t < 1024; t += 256) {
        int x = 1 + (t >> 6), d = t & 63;
        float run = 0.0f;
        for (int y = 1; y <= 16; ++y) {
            run += dst[(y * 17 + x) * 64 + d];
            dst[(y * 17 + x) * 64 + d] = run;
        }
    }
}

// ---------------------------------------------------------------------------
// 4. Fused area attention. 128 threads (4 waves) per (bh, 64-query tile).
//    Integral images cached in LDS (2 x 74 KB of the 320 KB WGP LDS),
//    moved with async b128 copies when available.
// ---------------------------------------------------------------------------
__global__ __launch_bounds__(128)
void area_attn_kernel(const float* __restrict__ qh,
                      const float* __restrict__ Ik,
                      const float* __restrict__ Iv,
                      _Float16* __restrict__ attn_h) {
    extern __shared__ __align__(16) char smem[];
    float*    s_Ik = (float*)smem;                 // 18496 f32
    float*    s_Iv = s_Ik + ISZ;                   // 18496 f32
    _Float16* s_K  = (_Float16*)(s_Iv + ISZ);      // [32 areas][64 dims]
    _Float16* s_VT = s_K + 32 * 64;                // [64 dims][32 areas]
    _Float16* s_P  = s_VT + 64 * 32;               // per-wave [16 q][32 a]

    const int qtile = blockIdx.x;   // 0..3
    const int bh    = blockIdx.y;   // 0..191
    const int tid   = threadIdx.x;
    const int lane  = tid & 31;
    const int wave  = tid >> 5;
    const int hf    = lane >> 4;
    const int ln    = lane & 15;

    const int qbase = qtile * 64 + wave * 16;
    const float* qp = qh + ((long)bh * NTOK + qbase + ln) * DHEAD;
    __builtin_prefetch(qp, 0, 1);     // global_prefetch_b8

    // --- cache integral images in LDS ---
    {
        const float* gIk = Ik + (long)bh * ISZ;
        const float* gIv = Iv + (long)bh * ISZ;
#if HAVE_ASYNC_LDS
        const uint4* g4k = (const uint4*)gIk;
        const uint4* g4v = (const uint4*)gIv;
        uint4* l4k = (uint4*)s_Ik;
        uint4* l4v = (uint4*)s_Iv;
        for (int i = tid; i < ISZ / 4; i += 128) {
            async_b128(g4k + i, l4k + i);
            async_b128(g4v + i, l4v + i);
        }
        ASYNC_WAIT(0);
#else
        for (int i = tid; i < ISZ; i += 128) { s_Ik[i] = gIk[i]; s_Iv[i] = gIv[i]; }
#endif
    }

    // --- Q fragments: 16 queries x 64 dims -> two 16x32 A fragments ---
    Frag16 qf[2];
#pragma unroll
    for (int ch = 0; ch < 2; ++ch)
#pragma unroll
        for (int i = 0; i < 8; ++i) {
            int kk = ch * 32 + kbaseA(i) + 8 * hf;
            qf[ch].h[2 * i + 0] = (_Float16)qp[kk + 0];
            qf[ch].h[2 * i + 1] = (_Float16)qp[kk + 1];
        }

    v8f accO[4];
#pragma unroll
    for (int c = 0; c < 4; ++c)
#pragma unroll
        for (int r = 0; r < 8; ++r) accO[c][r] = 0.0f;
    float mrow[8], lrow[8];
#pragma unroll
    for (int r = 0; r < 8; ++r) { mrow[r] = -1e30f; lrow[r] = 0.0f; }

    const int pa  = tid >> 2;          // producer: area 0..31
    const int pd0 = (tid & 3) * 16;    // producer: dim start

    __syncthreads();

    for (int step = 0; step < 64; ++step) {
        const int astart = step * 32;

        // --- produce K-area (mean) / V-area (sum) tiles from the SATs ---
        {
            int g = astart + pa;
            if (g < NAREA) {
                int ah = 1, aw = 1, py = 0, px = 0, off = 0;
#pragma unroll
                for (int seg = 0; seg < 9; ++seg) {
                    int th = seg / 3 + 1, tw = seg % 3 + 1;
                    int nh = 17 - th, nw = 17 - tw;
                    int cnt = nh * nw;
                    if (g >= off && g < off + cnt) {
                        ah = th; aw = tw;
                        py = (g - off) / nw; px = (g - off) % nw;
                    }
                    off += cnt;
                }
                float inv = 1.0f / (float)(ah * aw);
                int i11 = ((py + ah) * 17 + (px + aw)) * 64;
                int i01 = ((py     ) * 17 + (px + aw)) * 64;
                int i10 = ((py + ah) * 17 + (px     )) * 64;
                int i00 = ((py     ) * 17 + (px     )) * 64;
#pragma unroll 4
                for (int d = pd0; d < pd0 + 16; ++d) {
                    float ks = s_Ik[i11 + d] - s_Ik[i01 + d] - s_Ik[i10 + d] + s_Ik[i00 + d];
                    float vs = s_Iv[i11 + d] - s_Iv[i01 + d] - s_Iv[i10 + d] + s_Iv[i00 + d];
                    s_K[pa * 64 + d]  = (_Float16)(ks * inv);
                    s_VT[d * 32 + pa] = (_Float16)vs;
                }
            } else {
                for (int d = pd0; d < pd0 + 16; ++d) {
                    s_K[pa * 64 + d]  = (_Float16)0.0f;
                    s_VT[d * 32 + pa] = (_Float16)0.0f;
                }
            }
        }
        __syncthreads();

        // --- logits S[t] = Q (16x64) . K_area[t]^T (64x16) ---
        v8f S[2];
#pragma unroll
        for (int t = 0; t < 2; ++t) {
#pragma unroll
            for (int r = 0; r < 8; ++r) S[t][r] = 0.0f;
#pragma unroll
            for (int kc = 0; kc < 2; ++kc) {
                Frag16 bf;
                const int a = t * 16 + ln;
#pragma unroll
                for (int i = 0; i < 8; ++i) {
                    int kk = kc * 32 + kbaseA(i) + 8 * hf;
                    bf.u[i] = *(const unsigned int*)&s_K[a * 64 + kk];
                }
                S[t] = wmma_f16(qf[kc].v, bf.v, S[t]);
            }
            if (astart + t * 16 + ln >= NAREA) {
#pragma unroll
                for (int r = 0; r < 8; ++r) S[t][r] = -1e30f;
            }
        }

        // --- online softmax (row = query; 16 lanes per row group) ---
        float p0[8], p1[8];
#pragma unroll
        for (int r = 0; r < 8; ++r) {
            float mx = rowmax16(fmaxf(S[0][r], S[1][r]));
            float mnew = fmaxf(mrow[r], mx);
            float e0 = __expf(S[0][r] - mnew);
            float e1 = __expf(S[1][r] - mnew);
            float rs = rowsum16(e0 + e1);
            float scale = __expf(mrow[r] - mnew);
            lrow[r] = lrow[r] * scale + rs;
            mrow[r] = mnew;
#pragma unroll
            for (int c = 0; c < 4; ++c) accO[c][r] *= scale;
            p0[r] = e0; p1[r] = e1;
        }

        // --- stage P through per-wave LDS -> WMMA A fragment ---
        _Float16* pw = s_P + wave * (16 * 32);
#pragma unroll
        for (int r = 0; r < 8; ++r) {
            int mr = r + 8 * hf;
            pw[mr * 32 + ln]      = (_Float16)p0[r];
            pw[mr * 32 + 16 + ln] = (_Float16)p1[r];
        }
        Frag16 pf;
#pragma unroll
        for (int i = 0; i < 8; ++i) {
            int kk = kbaseA(i) + 8 * hf;
            pf.u[i] = *(const unsigned int*)&pw[ln * 32 + kk];
        }

        // --- O += P (16x32) @ V_area (32x64) ---
#pragma unroll
        for (int c = 0; c < 4; ++c) {
            Frag16 vf;
            const int dcol = c * 16 + ln;
#pragma unroll
            for (int i = 0; i < 8; ++i) {
                int kk = kbaseA(i) + 8 * hf;
                vf.u[i] = *(const unsigned int*)&s_VT[dcol * 32 + kk];
            }
            accO[c] = wmma_f16(pf.v, vf.v, accO[c]);
        }
        __syncthreads();
    }

    // --- epilogue: normalize, write f16 [B, N, H*64] ---
    const int b = bh / HEADS, h = bh % HEADS;
#pragma unroll
    for (int c = 0; c < 4; ++c)
#pragma unroll
        for (int r = 0; r < 8; ++r) {
            int nt  = qbase + r + 8 * hf;
            int col = h * 64 + c * 16 + ln;
            attn_h[(long)(b * NTOK + nt) * DIMX + col] =
                (_Float16)(accO[c][r] / lrow[r]);
        }
}

// ---------------------------------------------------------------------------
// Launch
// ---------------------------------------------------------------------------
extern "C" void kernel_launch(void* const* d_in, const int* in_sizes, int n_in,
                              void* d_out, int out_size, void* d_ws, size_t ws_size,
                              hipStream_t stream) {
    (void)in_sizes; (void)n_in; (void)out_size; (void)ws_size;
    const float* x     = (const float*)d_in[0];
    const float* w_qkv = (const float*)d_in[1];
    const float* w_q   = (const float*)d_in[2];
    const float* b_q   = (const float*)d_in[3];
    const float* w_k   = (const float*)d_in[4];
    const float* b_k   = (const float*)d_in[5];
    const float* w_v   = (const float*)d_in[6];
    const float* b_v   = (const float*)d_in[7];
    const float* w_o   = (const float*)d_in[8];
    const float* b_o   = (const float*)d_in[9];
    float* out = (float*)d_out;

    char* base = (char*)d_ws;
    _Float16* WcT    = (_Float16*)base; base += 3L * DIMX * DIMX * 2;       // 3.5 MB
    _Float16* WoT    = (_Float16*)base; base += 1L * DIMX * DIMX * 2;       // 1.2 MB
    _Float16* x_h    = (_Float16*)base; base += (long)MROWS * DIMX * 2;     // 6.3 MB
    _Float16* attn_h = (_Float16*)base; base += (long)MROWS * DIMX * 2;     // 6.3 MB
    float* qh  = (float*)base; base += (long)BHX * NTOK * DHEAD * 4;        // 12.6 MB
    float* kh  = (float*)base; base += (long)BHX * NTOK * DHEAD * 4;
    float* vh  = (float*)base; base += (long)BHX * NTOK * DHEAD * 4;
    float* Ikw = (float*)base; base += (long)BHX * ISZ * 4;                 // 14.2 MB
    float* Ivw = (float*)base; base += (long)BHX * ISZ * 4;

    // 0. precision conversion / transposition
    convert_f16_kernel<<<(MROWS * DIMX / 4 + 255) / 256, 256, 0, stream>>>(
        x, x_h, MROWS * DIMX / 4);
    transpose_f16_kernel<<<(DIMX * DIMX + 255) / 256, 256, 0, stream>>>(w_o, WoT);

    // 1. fold first projection into per-stream weights (transposed f16)
    combine_kernel<<<3 * DIMX * DIMX / 256, 256, 0, stream>>>(
        w_qkv, w_q, w_k, w_v, WcT);

    // 2. head projections
    dim3 gg(DIMX / 64, MROWS / 64);   // (12, 64)
    gemm_wmma_kernel<0><<<gg, 128, 0, stream>>>(x_h, WcT,                    b_q, qh);
    gemm_wmma_kernel<0><<<gg, 128, 0, stream>>>(x_h, WcT + 1L * DIMX * DIMX, b_k, kh);
    gemm_wmma_kernel<0><<<gg, 128, 0, stream>>>(x_h, WcT + 2L * DIMX * DIMX, b_v, vh);

    // 3. summed-area tables
    integral_kernel<<<dim3(BHX, 2), 256, 0, stream>>>(kh, vh, Ikw, Ivw);

    // 4. fused area attention
    size_t smem = (size_t)(2 * ISZ * sizeof(float) +
                           (32 * 64 + 64 * 32 + 4 * 16 * 32) * sizeof(_Float16));
    area_attn_kernel<<<dim3(4, BHX), 128, smem, stream>>>(qh, Ikw, Ivw, attn_h);

    // 5. output projection
    gemm_wmma_kernel<1><<<gg, 128, 0, stream>>>(attn_h, WoT, b_o, out);
}
